// NeighborAggPrefix_23072564314582
// MI455X (gfx1250) — matrix operand order repaired and checked by reading, not compiled, not run on hardware
//
#include <hip/hip_runtime.h>
#include <hip/hip_bf16.h>
#include <math.h>

// ---------- types ----------
typedef __attribute__((ext_vector_type(16))) __bf16          v16bf;
typedef __attribute__((ext_vector_type(8)))  float           v8f;
typedef __attribute__((ext_vector_type(8)))  unsigned short  v8u;

struct BfPair { v8u lo, hi; };

#define D_Z    256
#define D_PAIR 256
#define D_HID  3072     // 4*D_LM
#define D_OUT  12288    // M_PREFIX*D_LM

static __device__ __forceinline__ unsigned short f32_to_bf16_rne(float f) {
    unsigned int x = __builtin_bit_cast(unsigned int, f);
    unsigned int r = x + 0x7FFFu + ((x >> 16) & 1u);
    return (unsigned short)(r >> 16);
}

// ---------- bulk f32 -> bf16 conversion (pairs packed into u32) ----------
__global__ __launch_bounds__(256) void k_cvt_bf16(const float* __restrict__ src,
                                                  unsigned int* __restrict__ dst,
                                                  int n2) {       // n2 = n/2
    int i = blockIdx.x * 256 + threadIdx.x;
    if (i >= n2) return;
    float x0 = src[2 * i];
    float x1 = src[2 * i + 1];
    unsigned int lo = f32_to_bf16_rne(x0);
    unsigned int hi = f32_to_bf16_rne(x1);
    dst[i] = (hi << 16) | lo;
}

// A-fragment (16x32 bf16) from bf16 row-major [rows, ld]
static __device__ __forceinline__ v16bf load_frag_a_bf16(const unsigned short* base, int ld,
                                                         int row, int k, int lane) {
    int half = lane >> 4;
    int r    = row + (lane & 15);
    int k0   = k + half * 8;
    const unsigned short* p = base + (size_t)r * ld + k0;
    BfPair pr;
    pr.lo = *(const v8u*)(p);                 // K = k0..k0+7    -> v0..v3
    pr.hi = *(const v8u*)(p + 16);            // K = k0+16..+23  -> v4..v7
    return __builtin_bit_cast(v16bf, pr);
}

// B-fragment (32x16 bf16): column n = row n of weight matrix (K-contiguous)
static __device__ __forceinline__ v16bf load_frag_b_bf16(const unsigned short* base, int ld,
                                                         int col, int k, int lane) {
    int half = lane >> 4;
    int c    = col + (lane & 15);
    int k0   = k + half * 8;
    const unsigned short* p = base + (size_t)c * ld + k0;
    BfPair pr;
    pr.lo = *(const v8u*)(p);
    pr.hi = *(const v8u*)(p + 16);
    return __builtin_bit_cast(v16bf, pr);
}

// ---------- stage 1: C1 = Wv^T @ Wu  (256x256) ----------
__global__ __launch_bounds__(256) void k_wtw(const float* __restrict__ Wv,
                                             const float* __restrict__ Wu,
                                             float* __restrict__ C1) {
    int j = blockIdx.x * 16 + (threadIdx.x & 15);
    int i = blockIdx.y * 16 + (threadIdx.x >> 4);
    float s = 0.f;
    for (int h = 0; h < D_Z; ++h)
        s += Wv[h * D_Z + i] * Wu[h * D_Z + j];
    C1[i * D_Z + j] = s;
}

// ---------- stage 2: R = Z_self @ C1  ([B,256]) ----------
__global__ __launch_bounds__(256) void k_projR(const float* __restrict__ Z,
                                               const float* __restrict__ C1,
                                               float* __restrict__ R) {
    int idx = blockIdx.x * 256 + threadIdx.x;
    int m = idx >> 8, j = idx & 255;
    const float* z = Z + (size_t)m * D_Z;
    float s = 0.f;
    for (int kk = 0; kk < D_Z; ++kk)
        s += z[kk] * C1[kk * D_Z + j];
    R[idx] = s;
}

// ---------- stage 3: edge logits e_i = z_i . R[seg(i)] / 16 ----------
__global__ __launch_bounds__(256) void k_edge(const float* __restrict__ Zn,
                                              const float* __restrict__ R,
                                              const int* __restrict__ ptr,
                                              float* __restrict__ e,
                                              int E, int B) {
    int wave = threadIdx.x >> 5;
    int lane = threadIdx.x & 31;
    int i = blockIdx.x * 8 + wave;
    if (i >= E) return;
    // searchsorted(ptr, i, 'right') - 1 : largest j with ptr[j] <= i
    int lo = 0, hi = B;
    while (lo < hi) {
        int mid = (lo + hi + 1) >> 1;
        if (ptr[mid] <= i) lo = mid; else hi = mid - 1;
    }
    int seg = lo < (B - 1) ? lo : (B - 1);
    const float* z = Zn + (size_t)i * D_Z + lane * 8;
    const float* r = R  + (size_t)seg * D_Z + lane * 8;
    float s = 0.f;
#pragma unroll
    for (int t = 0; t < 8; ++t) s += z[t] * r[t];
#pragma unroll
    for (int off = 16; off; off >>= 1) s += __shfl_xor(s, off, 32);
    if (lane == 0) e[i] = s * 0.0625f;        // 1/sqrt(256)
}

// ---------- stage 4: per-segment softmax + weighted aggregation -> bf16 ----------
__global__ __launch_bounds__(256) void k_segagg(const float* __restrict__ e,
                                                const float* __restrict__ Ep,
                                                const int* __restrict__ ptr,
                                                unsigned short* __restrict__ EvXb) {
    __shared__ float red[256];
    __shared__ float abuf[256];
    int b = blockIdx.x, t = threadIdx.x;
    int start = ptr[b], end = ptr[b + 1];
    if (end <= start) { EvXb[(size_t)b * D_PAIR + t] = 0; return; }

    float m = -3.402823466e38f;
    for (int i = start + t; i < end; i += 256) m = fmaxf(m, e[i]);
    red[t] = m; __syncthreads();
#pragma unroll
    for (int s2 = 128; s2; s2 >>= 1) {
        if (t < s2) red[t] = fmaxf(red[t], red[t + s2]);
        __syncthreads();
    }
    float mx = red[0]; __syncthreads();

    float s = 0.f;
    for (int i = start + t; i < end; i += 256) s += expf(e[i] - mx);
    red[t] = s; __syncthreads();
#pragma unroll
    for (int s2 = 128; s2; s2 >>= 1) {
        if (t < s2) red[t] += red[t + s2];
        __syncthreads();
    }
    float inv = 1.0f / red[0]; __syncthreads();

    float acc = 0.f;
    for (int base = start; base < end; base += 256) {
        int cnt = min(256, end - base);
        if (t < cnt) abuf[t] = expf(e[base + t] - mx) * inv;
        __syncthreads();
        for (int j = 0; j < cnt; ++j)
            acc += abuf[j] * Ep[(size_t)(base + j) * D_PAIR + t];
        __syncthreads();
    }
    EvXb[(size_t)b * D_PAIR + t] = f32_to_bf16_rne(acc);
}

// ---------- ping-pong pipelined bf16 WMMA GEMM core (per-wave 32x64 tile) ----------
// acc[2][4] updated in-place; A: [M,K] bf16 row-major; W: [N,K] bf16 row-major.
// Requires K % 64 == 0.
static __device__ __forceinline__ void wmma_tile(const v16bf a[2], const v16bf b[4],
                                                 v8f acc[2][4]) {
#pragma unroll
    for (int ni = 0; ni < 4; ++ni) {
        acc[0][ni] = __builtin_amdgcn_wmma_f32_16x16x32_bf16(false, a[0], false, b[ni],
                                                             (short)0, acc[0][ni], false, false);
        acc[1][ni] = __builtin_amdgcn_wmma_f32_16x16x32_bf16(false, a[1], false, b[ni],
                                                             (short)0, acc[1][ni], false, false);
    }
}

static __device__ __forceinline__ void load_tile(const unsigned short* __restrict__ A,
                                                 const unsigned short* __restrict__ W,
                                                 int K, int Mb, int Nb, int k, int lane,
                                                 v16bf a[2], v16bf b[4]) {
    a[0] = load_frag_a_bf16(A, K, Mb,      k, lane);
    a[1] = load_frag_a_bf16(A, K, Mb + 16, k, lane);
#pragma unroll
    for (int ni = 0; ni < 4; ++ni)
        b[ni] = load_frag_b_bf16(W, K, Nb + ni * 16, k, lane);
}

static __device__ __forceinline__ void gemm_core(const unsigned short* __restrict__ A,
                                                 const unsigned short* __restrict__ W,
                                                 int K, int Mb, int Nb, int lane,
                                                 v8f acc[2][4]) {
    v16bf a0[2], b0[4], a1[2], b1[4];
    load_tile(A, W, K, Mb, Nb, 0, lane, a0, b0);

    for (int k = 0; k < K; k += 64) {
        // prefetch k+32 into buffer 1, consume buffer 0 (k)
        load_tile(A, W, K, Mb, Nb, k + 32, lane, a1, b1);
        wmma_tile(a0, b0, acc);
        // prefetch k+64 into buffer 0 (clamped on last iter; result discarded),
        // consume buffer 1 (k+32)
        int k2 = (k + 64 < K) ? (k + 64) : (k + 32);
        load_tile(A, W, K, Mb, Nb, k2, lane, a0, b0);
        wmma_tile(a1, b1, acc);
    }
}

// ---------- stage 5: hdn = gelu(EvX @ W1^T + b1) -> bf16 ----------
__global__ __launch_bounds__(256) void k_gemm1_gelu(const unsigned short* __restrict__ A,  // [M,256] bf16
                                                    const unsigned short* __restrict__ W1, // [3072,256] bf16
                                                    const float* __restrict__ b1,
                                                    unsigned short* __restrict__ H) {      // [M,3072] bf16
    int lane = threadIdx.x & 31;
    int wave = threadIdx.x >> 5;
    int wm = wave >> 2, wn = wave & 3;        // waves: 2 (M) x 4 (N)
    int Mb = blockIdx.x * 64 + wm * 32;
    int Nb = blockIdx.y * 256 + wn * 64;

    v8f acc[2][4];
#pragma unroll
    for (int mi = 0; mi < 2; ++mi)
#pragma unroll
        for (int ni = 0; ni < 4; ++ni)
            acc[mi][ni] = (v8f){0.f, 0.f, 0.f, 0.f, 0.f, 0.f, 0.f, 0.f};

    gemm_core(A, W1, D_Z, Mb, Nb, lane, acc);

    int colL  = lane & 15;
    int rhalf = (lane >> 4) * 8;
#pragma unroll
    for (int mi = 0; mi < 2; ++mi)
#pragma unroll
        for (int ni = 0; ni < 4; ++ni) {
            int col = Nb + ni * 16 + colL;
            float bias = b1[col];
#pragma unroll
            for (int j = 0; j < 8; ++j) {
                int row = Mb + mi * 16 + rhalf + j;
                float x = acc[mi][ni][j] + bias;
                float g = 0.5f * x * (1.0f + erff(x * 0.7071067811865476f));
                H[(size_t)row * D_HID + col] = f32_to_bf16_rne(g);
            }
        }
}

// ---------- stage 6: P = hdn @ W2^T + b2 -> f32 out ----------
__global__ __launch_bounds__(256) void k_gemm2(const unsigned short* __restrict__ H,  // [M,3072] bf16
                                               const unsigned short* __restrict__ W2, // [12288,3072] bf16
                                               const float* __restrict__ b2,
                                               float* __restrict__ P) {               // [M,12288]
    int lane = threadIdx.x & 31;
    int wave = threadIdx.x >> 5;
    int wm = wave >> 2, wn = wave & 3;
    int Mb = blockIdx.x * 64 + wm * 32;
    int Nb = blockIdx.y * 256 + wn * 64;

    v8f acc[2][4];
#pragma unroll
    for (int mi = 0; mi < 2; ++mi)
#pragma unroll
        for (int ni = 0; ni < 4; ++ni)
            acc[mi][ni] = (v8f){0.f, 0.f, 0.f, 0.f, 0.f, 0.f, 0.f, 0.f};

    gemm_core(H, W2, D_HID, Mb, Nb, lane, acc);

    int colL  = lane & 15;
    int rhalf = (lane >> 4) * 8;
#pragma unroll
    for (int mi = 0; mi < 2; ++mi)
#pragma unroll
        for (int ni = 0; ni < 4; ++ni) {
            int col = Nb + ni * 16 + colL;
            float bias = b2[col];
#pragma unroll
            for (int j = 0; j < 8; ++j) {
                int row = Mb + mi * 16 + rhalf + j;
                P[(size_t)row * D_OUT + col] = acc[mi][ni][j] + bias;
            }
        }
}

// ---------- launcher ----------
extern "C" void kernel_launch(void* const* d_in, const int* in_sizes, int n_in,
                              void* d_out, int out_size, void* d_ws, size_t ws_size,
                              hipStream_t stream) {
    const float* Zs  = (const float*)d_in[0];
    const float* Zn  = (const float*)d_in[1];
    const float* Ep  = (const float*)d_in[2];
    const float* Wv  = (const float*)d_in[3];
    const float* Wu  = (const float*)d_in[4];
    const float* W1  = (const float*)d_in[5];
    const float* b1  = (const float*)d_in[6];
    const float* W2  = (const float*)d_in[7];
    const float* b2  = (const float*)d_in[8];
    const int*   ptr = (const int*)d_in[9];

    const int B = in_sizes[0] / D_Z;      // 4096
    const int E = in_sizes[1] / D_Z;      // 131072
    const int n_w1 = in_sizes[5];         // 3072*256
    const int n_w2 = in_sizes[7];         // 12288*3072

    char* ws = (char*)d_ws;
    float* C1 = (float*)ws;                  ws += (size_t)D_Z * D_Z * 4;           // 256 KB
    float* R  = (float*)ws;                  ws += (size_t)B * D_Z * 4;             // 4 MB
    float* ev = (float*)ws;                  ws += (size_t)E * 4;                   // 512 KB
    unsigned short* EvXb = (unsigned short*)ws; ws += (size_t)B * D_PAIR * 2;       // 2 MB
    unsigned short* W1b  = (unsigned short*)ws; ws += (size_t)n_w1 * 2;             // 1.5 MB
    unsigned short* W2b  = (unsigned short*)ws; ws += (size_t)n_w2 * 2;             // 72 MB
    unsigned short* Hb   = (unsigned short*)ws;                                     // 24 MB

    // weight conversions (independent, bandwidth-bound)
    k_cvt_bf16<<<(n_w1 / 2 + 255) / 256, 256, 0, stream>>>(W1, (unsigned int*)W1b, n_w1 / 2);
    k_cvt_bf16<<<(n_w2 / 2 + 255) / 256, 256, 0, stream>>>(W2, (unsigned int*)W2b, n_w2 / 2);

    k_wtw   <<<dim3(16, 16), 256, 0, stream>>>(Wv, Wu, C1);
    k_projR <<<B, 256, 0, stream>>>(Zs, C1, R);
    k_edge  <<<(E + 7) / 8, 256, 0, stream>>>(Zn, R, ptr, ev, E, B);
    k_segagg<<<B, 256, 0, stream>>>(ev, Ep, ptr, EvXb);
    k_gemm1_gelu<<<dim3(B / 64, D_HID / 256), 256, 0, stream>>>(EvXb, W1b, b1, Hb);
    k_gemm2     <<<dim3(B / 64, D_OUT / 256), 256, 0, stream>>>(Hb, W2b, b2, (float*)d_out);
}
